// EncoderLayer_2817498546298
// MI455X (gfx1250) — matrix operand order repaired and checked by reading, not compiled, not run on hardware
//
#include <hip/hip_runtime.h>
#include <math.h>

// ---------------- problem constants ----------------
#define HIDN 1024
#define NHEAD 16
#define HDIM 64
#define SEQ 2048
#define FFD 4096
#define HWID 128
#define PPOS 257
#define PPAD 272

// ---------------- WMMA plumbing (CDNA5 / gfx1250, wave32) ----------------
typedef __attribute__((ext_vector_type(16))) __bf16 v16bf;
typedef __attribute__((ext_vector_type(8)))  float  v8f;
typedef __attribute__((ext_vector_type(4)))  int    v4i;

union FragU { v16bf v; unsigned int u[8]; };

__device__ __forceinline__ unsigned short f2bf(float f) {
  union { float f; unsigned int u; } x; x.f = f;
  unsigned int r = x.u + 0x7FFFu + ((x.u >> 16) & 1u);   // RNE
  return (unsigned short)(r >> 16);
}
__device__ __forceinline__ float bf2f(unsigned short b) {
  union { unsigned int u; float f; } x; x.u = ((unsigned int)b) << 16;
  return x.f;
}

__device__ __forceinline__ v8f wmma_bf16(v16bf a, v16bf b, v8f c) {
  return __builtin_amdgcn_wmma_f32_16x16x32_bf16(false, a, false, b, (short)0, c, false, false);
}

// A-matrix 16x32 bf16 layout (ISA 7.12.2): lanes 0-15: v0-3 -> K=0..7, v4-7 -> K=16..23
//                                          lanes16-31: v0-3 -> K=8..15, v4-7 -> K=24..31
__device__ __forceinline__ v16bf load_fragA(const unsigned short* rowptr) {
  const int hh = (threadIdx.x >> 4) & 1;
  const unsigned int* p = (const unsigned int*)rowptr;
  FragU f;
#pragma unroll
  for (int v = 0; v < 8; ++v) {
    const int kb = ((v >> 2) << 4) + (hh << 3) + ((v & 3) << 1);
    f.u[v] = p[kb >> 1];
  }
  return f.v;
}

// B-matrix 32x16 bf16 layout: lanes 0-15: v0-3 -> K=0..7, v4-7 -> K=16..23; lanes 16-31: K=8..15, 24..31.
__device__ __forceinline__ v16bf load_fragB(const unsigned short* rowptr) {
  const int hh = (threadIdx.x >> 4) & 1;
  const unsigned int* p = (const unsigned int*)rowptr + (hh << 2);
  FragU f;
#pragma unroll
  for (int v = 0; v < 8; ++v) f.u[v] = p[((v >> 2) << 3) + (v & 3)];
  return f.v;
}

// ---------------- CDNA5 async global->LDS (ASYNCcnt path) ----------------
typedef __attribute__((address_space(3))) v4i* lptr_v4i;

__device__ __forceinline__ void async_load_b128(const unsigned short* g, unsigned lds_off) {
#if __has_builtin(__builtin_amdgcn_global_load_async_to_lds_b128)
  __builtin_amdgcn_global_load_async_to_lds_b128(
      (v4i*)(uintptr_t)g, (lptr_v4i)(uintptr_t)lds_off, 0, 0);
#else
  asm volatile("global_load_async_to_lds_b128 %0, %1, off"
               :: "v"(lds_off), "v"((unsigned long long)(uintptr_t)g)
               : "memory");
#endif
}

template <int N>
__device__ __forceinline__ void wait_async() {
#if __has_builtin(__builtin_amdgcn_s_wait_asynccnt)
  __builtin_amdgcn_s_wait_asynccnt(N);
#else
  asm volatile("s_wait_asynccnt %0" :: "n"(N) : "memory");
#endif
}

// ---------------- transpose + fp32->bf16 convert (weights, pos tables) ----------------
// dst[b][c][r] = bf16(src[b][r][c]);  dst per-batch is [cols][rows] (row stride = rows)
__global__ void transpose_cvt_kernel(const float* __restrict__ src,
                                     unsigned short* __restrict__ dst,
                                     int rows, int cols,
                                     long long sbatch, long long dbatch) {
  __shared__ float tile[32][33];
  const float* s = src + (long long)blockIdx.z * sbatch;
  unsigned short* d = dst + (long long)blockIdx.z * dbatch;
  const int c0 = blockIdx.x * 32, r0 = blockIdx.y * 32;
  for (int i = threadIdx.y; i < 32; i += 8) {
    int r = r0 + i, c = c0 + threadIdx.x;
    tile[i][threadIdx.x] = (r < rows && c < cols) ? s[(size_t)r * cols + c] : 0.f;
  }
  __syncthreads();
  for (int i = threadIdx.y; i < 32; i += 8) {
    int c = c0 + i, r = r0 + threadIdx.x;
    if (c < cols && r < rows) d[(size_t)c * rows + r] = f2bf(tile[threadIdx.x][i]);
  }
}

// bf16 [rows][cols] -> bf16 [cols][rows]   (V -> V^T)
__global__ void transpose_bf16_kernel(const unsigned short* __restrict__ src,
                                      unsigned short* __restrict__ dst,
                                      int rows, int cols) {
  __shared__ unsigned short tile[32][34];
  const int c0 = blockIdx.x * 32, r0 = blockIdx.y * 32;
  for (int i = threadIdx.y; i < 32; i += 8) {
    int r = r0 + i, c = c0 + threadIdx.x;
    tile[i][threadIdx.x] = (r < rows && c < cols) ? src[(size_t)r * cols + c] : (unsigned short)0;
  }
  __syncthreads();
  for (int i = threadIdx.y; i < 32; i += 8) {
    int c = c0 + i, r = r0 + threadIdx.x;
    if (c < cols && r < rows) dst[(size_t)c * rows + r] = tile[threadIdx.x][i];
  }
}

// ---------------- LayerNorm: fp32 in -> bf16 out ----------------
__global__ __launch_bounds__(256) void ln_kernel(const float* __restrict__ x,
                                                 const float* __restrict__ g,
                                                 const float* __restrict__ b,
                                                 unsigned short* __restrict__ out,
                                                 int cols) {
  __shared__ float red[256];
  const int row = blockIdx.x;
  const float* xr = x + (size_t)row * cols;
  float s = 0.f;
  for (int c = threadIdx.x; c < cols; c += 256) s += xr[c];
  red[threadIdx.x] = s; __syncthreads();
  for (int o = 128; o > 0; o >>= 1) { if (threadIdx.x < o) red[threadIdx.x] += red[threadIdx.x + o]; __syncthreads(); }
  const float mu = red[0] / (float)cols;
  __syncthreads();
  float vs = 0.f;
  for (int c = threadIdx.x; c < cols; c += 256) { float d = xr[c] - mu; vs += d * d; }
  red[threadIdx.x] = vs; __syncthreads();
  for (int o = 128; o > 0; o >>= 1) { if (threadIdx.x < o) red[threadIdx.x] += red[threadIdx.x + o]; __syncthreads(); }
  const float inv = rsqrtf(red[0] / (float)cols + 1e-8f);
  unsigned short* orow = out + (size_t)row * cols;
  for (int c = threadIdx.x; c < cols; c += 256)
    orow[c] = f2bf((xr[c] - mu) * inv * g[c] + b[c]);
}

// ---------------- bf16 WMMA GEMM with async-to-LDS B staging ----------------
// C[M][N] = act( A[M][K](bf16) * Bt[N][K](bf16) + bias[N] ) + resid[M][N]
// block = 256 threads (8 waves), macro-tile 128(M) x 64(N); wave tile 32x32.
// B tile (64 N-rows x 64 K) is double-buffered in LDS, filled with
// GLOBAL_LOAD_ASYNC_TO_LDS_B128 and synchronized via ASYNCcnt.
// A comes straight from global (L2-resident on MI455X's 192MB L2).
#define BSTR 72   // LDS row stride (elements): 36 dwords -> conflict-free b128 reads
__global__ __launch_bounds__(256) void gemm_bf16_kernel(
    const unsigned short* __restrict__ A, const unsigned short* __restrict__ Bt,
    const float* __restrict__ bias, const float* __restrict__ resid,
    float* __restrict__ outF, unsigned short* __restrict__ outB,
    int M, int N, int K, int gelu) {
  __shared__ unsigned short Bl[2][64][BSTR];

  const int tid = threadIdx.x;
  const int lane = tid & 31;
  const int wave = tid >> 5;
  const int m0 = blockIdx.x * 128 + (wave & 3) * 32;
  const int nw = (wave >> 2) * 32;               // wave's N offset inside the 64-wide tile
  const int n0 = blockIdx.y * 64 + nw;           // global N base for epilogue

  // async-fill mapping: 512 b128 chunks (64 rows x 8 chunks), 2 per thread
  const int rr = tid >> 3;                       // 0..31
  const int cc = (tid & 7) << 3;                 // element offset of 16B chunk
  const unsigned short* gsrc0 = Bt + (size_t)(blockIdx.y * 64 + rr) * K + cc;
  const unsigned short* gsrc1 = Bt + (size_t)(blockIdx.y * 64 + rr + 32) * K + cc;

  v8f acc[2][2] = {};
  const unsigned short* arow0 = A + (size_t)(m0 + (lane & 15)) * K;
  const unsigned short* arow1 = arow0 + (size_t)16 * K;
  const unsigned short* brow0 = &Bl[0][nw + (lane & 15)][0];
  const unsigned short* brow1 = brow0 + 16 * BSTR;
  const unsigned bufstride = (unsigned)(64 * BSTR * 2);

  const unsigned ldst0 = (unsigned)(uintptr_t)&Bl[0][rr][cc];
  const unsigned ldst1 = (unsigned)(uintptr_t)&Bl[0][rr + 32][cc];

  // prefetch first two 64-K batches
  async_load_b128(gsrc0, ldst0);
  async_load_b128(gsrc1, ldst1);
  async_load_b128(gsrc0 + 64, ldst0 + bufstride);
  async_load_b128(gsrc1 + 64, ldst1 + bufstride);

  const int NT = K >> 6;
  for (int it = 0; it < NT; ++it) {
    if (it + 1 < NT) wait_async<2>(); else wait_async<0>();
    __syncthreads();                             // everyone's chunks have landed

    const int k0 = it << 6;
    const int cur = it & 1;
    const unsigned short* b0base = brow0 + cur * 64 * BSTR;
    const unsigned short* b1base = brow1 + cur * 64 * BSTR;
#pragma unroll
    for (int ks = 0; ks < 64; ks += 32) {
      v16bf a0 = load_fragA(arow0 + k0 + ks);
      v16bf a1 = load_fragA(arow1 + k0 + ks);
      v16bf b0 = load_fragB(b0base + ks);
      v16bf b1 = load_fragB(b1base + ks);
      acc[0][0] = wmma_bf16(a0, b0, acc[0][0]);
      acc[0][1] = wmma_bf16(a0, b1, acc[0][1]);
      acc[1][0] = wmma_bf16(a1, b0, acc[1][0]);
      acc[1][1] = wmma_bf16(a1, b1, acc[1][1]);
    }
    __syncthreads();                             // all waves done reading buffer `cur`
    if (it + 2 < NT) {                           // refill freed buffer for k0+128
      async_load_b128(gsrc0 + k0 + 128, ldst0 + (unsigned)cur * bufstride);
      async_load_b128(gsrc1 + k0 + 128, ldst1 + (unsigned)cur * bufstride);
    }
  }

  const int mr = (lane >> 4) * 8;
#pragma unroll
  for (int im = 0; im < 2; ++im)
#pragma unroll
    for (int in = 0; in < 2; ++in) {
      const int col = n0 + in * 16 + (lane & 15);
      const float bv = bias ? bias[col] : 0.f;
#pragma unroll
      for (int t = 0; t < 8; ++t) {
        const int row = m0 + im * 16 + mr + t;
        float v = acc[im][in][t] + bv;
        if (gelu) v = 0.5f * v * (1.f + erff(v * 0.70710678118654752f));  // exact GELU
        if (resid) v += resid[(size_t)row * N + col];
        if (outF) outF[(size_t)row * N + col] = v;
        else      outB[(size_t)row * N + col] = f2bf(v);
      }
    }
}

// ---------------- pos-bias tables: C2P[h][i][p]=q_i.PK[:,p], P2C[h][j][p]=k_j.PQ[:,p] ----------------
__global__ __launch_bounds__(32) void pos_gemm_kernel(
    const unsigned short* __restrict__ q, const unsigned short* __restrict__ k,
    const unsigned short* __restrict__ pkt, const unsigned short* __restrict__ pqt,
    unsigned short* __restrict__ c2p, unsigned short* __restrict__ p2c) {
  const int lane = threadIdx.x;
  const int head = blockIdx.z & 15;
  const unsigned short* X  = (blockIdx.z < 16) ? q   : k;
  const unsigned short* Pt = (blockIdx.z < 16) ? pkt : pqt;
  unsigned short*      Out = (blockIdx.z < 16) ? c2p : p2c;
  const int m0 = blockIdx.x * 16, p0 = blockIdx.y * 16;

  const unsigned short* ar = X  + (size_t)(m0 + (lane & 15)) * HIDN + head * HDIM;
  const unsigned short* br = Pt + ((size_t)head * PPAD + p0 + (lane & 15)) * HDIM;
  v8f acc = {};
  acc = wmma_bf16(load_fragA(ar),      load_fragB(br),      acc);
  acc = wmma_bf16(load_fragA(ar + 32), load_fragB(br + 32), acc);

  const int col = p0 + (lane & 15);
  const int mr = (lane >> 4) * 8;
#pragma unroll
  for (int t = 0; t < 8; ++t) {
    const int row = m0 + mr + t;
    Out[((size_t)head * SEQ + row) * PPAD + col] = f2bf(acc[t]);
  }
}

// ---------------- banded disentangled attention ----------------
#define SSTR 292
__global__ __launch_bounds__(32) void attn_kernel(
    const unsigned short* __restrict__ qb, const unsigned short* __restrict__ kb,
    const unsigned short* __restrict__ vt,   // [HIDN][SEQ] bf16 (V transposed)
    const unsigned short* __restrict__ c2p, const unsigned short* __restrict__ p2c,
    const unsigned char* __restrict__ amask,
    unsigned short* __restrict__ ctx) {
  __shared__ float S[16 * SSTR];
  __shared__ unsigned short P[16 * SSTR];
  __shared__ float red[32];

  const int lane = threadIdx.x;
  const int qblk = blockIdx.x, head = blockIdx.y;
  const int i0 = qblk * 16;
  const int base = i0 - 128 - ((qblk & 1) ? 16 : 0);   // 32-aligned window start
  const int nlo = lane & 15, mr = (lane >> 4) * 8;

  const unsigned short* qrow = qb + (size_t)(i0 + nlo) * HIDN + head * HDIM;
  const v16bf aq0 = load_fragA(qrow);
  const v16bf aq1 = load_fragA(qrow + 32);

  // ---- phase 1: scores ----
  for (int kt = 0; kt < 18; ++kt) {
    const int j0 = base + kt * 16;
    const int jl = j0 + nlo;
    const int jc = jl < 0 ? 0 : (jl > SEQ - 1 ? SEQ - 1 : jl);
    const unsigned short* krow = kb + (size_t)jc * HIDN + head * HDIM;
    v8f acc = {};
    acc = wmma_bf16(aq0, load_fragB(krow),      acc);
    acc = wmma_bf16(aq1, load_fragB(krow + 32), acc);
    const bool jok = (jl >= 0) && (jl < SEQ) && (amask[jl] != 0);
#pragma unroll
    for (int t = 0; t < 8; ++t) {
      const int i = i0 + mr + t;
      const int d = jl - i;
      float val = -3.0e38f;
      if (jok && d >= -HWID && d <= HWID && amask[i]) {
        const float bias =
            bf2f(c2p[((size_t)head * SEQ + i ) * PPAD + (HWID + d)]) +
            bf2f(p2c[((size_t)head * SEQ + jl) * PPAD + (HWID - d)]);
        val = acc[t] + bias;
      }
      S[(mr + t) * SSTR + kt * 16 + nlo] = val;
    }
  }
  __syncthreads();

  // ---- phase 2: softmax (2 lanes per row, 144 cols each) ----
  {
    const int r = lane & 15, hh = lane >> 4;
    const int c0 = hh * 144;
    float mx = -3.4e38f;
    for (int c = 0; c < 144; ++c) mx = fmaxf(mx, S[r * SSTR + c0 + c]);
    red[lane] = mx; __syncthreads();
    mx = fmaxf(red[r], red[r + 16]); __syncthreads();
    float sum = 0.f;
    for (int c = 0; c < 144; ++c) {
      float e = __expf(S[r * SSTR + c0 + c] - mx);
      S[r * SSTR + c0 + c] = e;
      sum += e;
    }
    red[lane] = sum; __syncthreads();
    const float inv = 1.f / (red[r] + red[r + 16]);
    for (int c = 0; c < 144; ++c)
      P[r * SSTR + c0 + c] = f2bf(S[r * SSTR + c0 + c] * inv);
  }
  __syncthreads();

  // ---- phase 3: ctx = P * V ----
  v8f acc[4] = {};
  const int hh = lane >> 4;
  for (int ks = 0; ks < 9; ++ks) {
    FragU fa;
    const unsigned int* prow = (const unsigned int*)&P[nlo * SSTR];
#pragma unroll
    for (int v = 0; v < 8; ++v) {
      const int kb2 = ((v >> 2) << 4) + (hh << 3) + ((v & 3) << 1);
      fa.u[v] = prow[(ks * 32 + kb2) >> 1];
    }
#pragma unroll
    for (int n = 0; n < 4; ++n) {
      const unsigned short* vrow = vt + (size_t)(head * HDIM + n * 16 + nlo) * SEQ;
      FragU fb;
#pragma unroll
      for (int v = 0; v < 8; ++v) {
        int jj = base + ks * 32 + ((v >> 2) << 4) + (hh << 3) + ((v & 3) << 1);   // B layout K index
        jj = jj < 0 ? 0 : (jj > SEQ - 2 ? SEQ - 2 : jj);  // clamp (P==0 there)
        fb.u[v] = *(const unsigned int*)(vrow + jj);
      }
      acc[n] = wmma_bf16(fa.v, fb.v, acc[n]);
    }
  }
#pragma unroll
  for (int n = 0; n < 4; ++n)
#pragma unroll
    for (int t = 0; t < 8; ++t) {
      const int i = i0 + mr + t;
      ctx[(size_t)i * HIDN + head * HDIM + n * 16 + nlo] = f2bf(acc[n][t]);
    }
}

// ---------------- host orchestration ----------------
extern "C" void kernel_launch(void* const* d_in, const int* in_sizes, int n_in,
                              void* d_out, int out_size, void* d_ws, size_t ws_size,
                              hipStream_t stream) {
  (void)in_sizes; (void)n_in; (void)out_size; (void)ws_size;
  const float* hidden          = (const float*)d_in[0];
  const unsigned char* amask   = (const unsigned char*)d_in[1];
  const float* wq  = (const float*)d_in[2];
  const float* wk  = (const float*)d_in[3];
  const float* wv  = (const float*)d_in[4];
  const float* pos_key   = (const float*)d_in[5];
  const float* pos_query = (const float*)d_in[6];
  const float* wo  = (const float*)d_in[7];
  const float* bo  = (const float*)d_in[8];
  const float* ln1_g = (const float*)d_in[9];
  const float* ln1_b = (const float*)d_in[10];
  const float* ln2_g = (const float*)d_in[11];
  const float* ln2_b = (const float*)d_in[12];
  const float* w1 = (const float*)d_in[13];
  const float* b1 = (const float*)d_in[14];
  const float* w2 = (const float*)d_in[15];
  const float* b2 = (const float*)d_in[16];

  char* ws = (char*)d_ws;
  size_t off = 0;
  auto take = [&](size_t bytes) -> void* {
    void* p = (void*)(ws + off);
    off = (off + bytes + 255) & ~(size_t)255;
    return p;
  };
  unsigned short* WQt = (unsigned short*)take((size_t)HIDN * HIDN * 2);
  unsigned short* WKt = (unsigned short*)take((size_t)HIDN * HIDN * 2);
  unsigned short* WVt = (unsigned short*)take((size_t)HIDN * HIDN * 2);
  unsigned short* WOt = (unsigned short*)take((size_t)HIDN * HIDN * 2);
  unsigned short* W1t = (unsigned short*)take((size_t)FFD * HIDN * 2);
  unsigned short* W2t = (unsigned short*)take((size_t)HIDN * FFD * 2);
  unsigned short* PKt = (unsigned short*)take((size_t)NHEAD * PPAD * HDIM * 2);
  unsigned short* PQt = (unsigned short*)take((size_t)NHEAD * PPAD * HDIM * 2);
  unsigned short* x1b = (unsigned short*)take((size_t)SEQ * HIDN * 2);
  unsigned short* qbuf = (unsigned short*)take((size_t)SEQ * HIDN * 2);
  unsigned short* kbuf = (unsigned short*)take((size_t)SEQ * HIDN * 2);
  unsigned short* vbuf = (unsigned short*)take((size_t)SEQ * HIDN * 2);
  unsigned short* vtb  = (unsigned short*)take((size_t)HIDN * SEQ * 2);
  unsigned short* C2P  = (unsigned short*)take((size_t)NHEAD * SEQ * PPAD * 2);
  unsigned short* P2C  = (unsigned short*)take((size_t)NHEAD * SEQ * PPAD * 2);
  unsigned short* ctxb = (unsigned short*)take((size_t)SEQ * HIDN * 2);
  float*          hbuf = (float*)take((size_t)SEQ * HIDN * 4);
  unsigned short* x2b  = (unsigned short*)take((size_t)SEQ * HIDN * 2);
  unsigned short* ff1b = (unsigned short*)take((size_t)SEQ * FFD * 2);

  const dim3 tpb(32, 8, 1);
  // weight prep: fp32 [K][N] -> bf16 [N][K]
  transpose_cvt_kernel<<<dim3(32, 32, 1), tpb, 0, stream>>>(wq, WQt, HIDN, HIDN, 0, 0);
  transpose_cvt_kernel<<<dim3(32, 32, 1), tpb, 0, stream>>>(wk, WKt, HIDN, HIDN, 0, 0);
  transpose_cvt_kernel<<<dim3(32, 32, 1), tpb, 0, stream>>>(wv, WVt, HIDN, HIDN, 0, 0);
  transpose_cvt_kernel<<<dim3(32, 32, 1), tpb, 0, stream>>>(wo, WOt, HIDN, HIDN, 0, 0);
  transpose_cvt_kernel<<<dim3(128, 32, 1), tpb, 0, stream>>>(w1, W1t, HIDN, FFD, 0, 0);
  transpose_cvt_kernel<<<dim3(32, 128, 1), tpb, 0, stream>>>(w2, W2t, FFD, HIDN, 0, 0);
  // pos tables: [H][D][257] -> [H][257(pad272)][D] bf16
  transpose_cvt_kernel<<<dim3(9, 2, NHEAD), tpb, 0, stream>>>(
      pos_key, PKt, HDIM, PPOS, (long long)HDIM * PPOS, (long long)PPAD * HDIM);
  transpose_cvt_kernel<<<dim3(9, 2, NHEAD), tpb, 0, stream>>>(
      pos_query, PQt, HDIM, PPOS, (long long)HDIM * PPOS, (long long)PPAD * HDIM);

  // LN1
  ln_kernel<<<SEQ, 256, 0, stream>>>(hidden, ln1_g, ln1_b, x1b, HIDN);

  // QKV projections (WMMA bf16, async-LDS staged B)
  gemm_bf16_kernel<<<dim3(16, 16), 256, 0, stream>>>(x1b, WQt, nullptr, nullptr, nullptr, qbuf, SEQ, HIDN, HIDN, 0);
  gemm_bf16_kernel<<<dim3(16, 16), 256, 0, stream>>>(x1b, WKt, nullptr, nullptr, nullptr, kbuf, SEQ, HIDN, HIDN, 0);
  gemm_bf16_kernel<<<dim3(16, 16), 256, 0, stream>>>(x1b, WVt, nullptr, nullptr, nullptr, vbuf, SEQ, HIDN, HIDN, 0);
  transpose_bf16_kernel<<<dim3(32, 64, 1), tpb, 0, stream>>>(vbuf, vtb, SEQ, HIDN);

  // relative-position bias tables (WMMA)
  pos_gemm_kernel<<<dim3(SEQ / 16, 17, 32), 32, 0, stream>>>(qbuf, kbuf, PKt, PQt, C2P, P2C);

  // banded attention + softmax + ctx (WMMA)
  attn_kernel<<<dim3(SEQ / 16, NHEAD), 32, 0, stream>>>(qbuf, kbuf, vtb, C2P, P2C, amask, ctxb);

  // output projection + residual: h = hidden + ctx@wo + bo
  gemm_bf16_kernel<<<dim3(16, 16), 256, 0, stream>>>(ctxb, WOt, bo, hidden, hbuf, nullptr, SEQ, HIDN, HIDN, 0);

  // LN2 + FFN
  ln_kernel<<<SEQ, 256, 0, stream>>>(hbuf, ln2_g, ln2_b, x2b, HIDN);
  gemm_bf16_kernel<<<dim3(16, 64), 256, 0, stream>>>(x2b, W1t, b1, nullptr, nullptr, ff1b, SEQ, FFD, HIDN, 1);
  gemm_bf16_kernel<<<dim3(16, 16), 256, 0, stream>>>(ff1b, W2t, b2, hbuf, (float*)d_out, nullptr, SEQ, HIDN, FFD, 0);
}
